// SNN_83554293776423
// MI455X (gfx1250) — compile-verified, matrix operand order checked by reading, not run on hardware
//
#include <hip/hip_runtime.h>
#include <hip/hip_bf16.h>

typedef __attribute__((ext_vector_type(16))) _Float16 v16h;
typedef __attribute__((ext_vector_type(8)))  _Float16 v8h;
typedef __attribute__((ext_vector_type(2)))  _Float16 v2h;
typedef __attribute__((ext_vector_type(8)))  float    v8f;

#define BM 128
#define BN 128
#define BK 32
#define LDA 40           // padded LDS row stride in halves (80B) to skew banks
#define BUFH (BM * LDA)  // halves per A (or B) buffer = 5120 (10240 B)

union Frag16 { v16h v; v8h h[2]; };

// CDNA5 async copy: global -> LDS, 16 bytes per lane, tracked by ASYNCcnt.
__device__ __forceinline__ void async_g2l_b128(unsigned lds_addr, const void* gptr)
{
    asm volatile("global_load_async_to_lds_b128 %0, %1, off"
                 :: "v"(lds_addr), "v"(gptr) : "memory");
}
// Wait ASYNCcnt <= 4: lets the 4 copies just issued stay in flight while
// guaranteeing (in-order completion) that the previous chunk's copies landed.
// acc operands are a scheduling fence so this cannot hoist above the WMMAs.
__device__ __forceinline__ void wait_async_le4(const v8f* a)
{
    asm volatile("s_wait_asynccnt 0x4"
                 :: "v"(a[0][0]), "v"(a[1][0]), "v"(a[2][0]), "v"(a[3][0]),
                    "v"(a[4][0]), "v"(a[5][0]), "v"(a[6][0]), "v"(a[7][0])
                 : "memory");
}
__device__ __forceinline__ void wait_async_le0(const v8f* a)
{
    asm volatile("s_wait_asynccnt 0x0"
                 :: "v"(a[0][0]), "v"(a[1][0]), "v"(a[2][0]), "v"(a[3][0]),
                    "v"(a[4][0]), "v"(a[5][0]), "v"(a[6][0]), "v"(a[7][0])
                 : "memory");
}

// ---------------------------------------------------------------------------
// f32 -> f16 conversion with zero-padding of the K dimension (784 % 8 == 0,
// so each 8-group is fully in or fully out of range).
// ---------------------------------------------------------------------------
__global__ __launch_bounds__(256)
void snn_cvt_f16_pad(const float* __restrict__ src, _Float16* __restrict__ dst,
                     int rows, int Kreal, int Kpad)
{
    const int gid = blockIdx.x * blockDim.x + threadIdx.x;
    const int kb  = Kpad >> 3;
    if (gid >= rows * kb) return;
    const int r  = gid / kb;
    const int k8 = (gid - r * kb) * 8;

    v8h out;
    if (k8 + 8 <= Kreal) {
        const float4 f0 = *(const float4*)(src + (size_t)r * Kreal + k8);
        const float4 f1 = *(const float4*)(src + (size_t)r * Kreal + k8 + 4);
        out[0] = (_Float16)f0.x; out[1] = (_Float16)f0.y;
        out[2] = (_Float16)f0.z; out[3] = (_Float16)f0.w;
        out[4] = (_Float16)f1.x; out[5] = (_Float16)f1.y;
        out[6] = (_Float16)f1.z; out[7] = (_Float16)f1.w;
    } else {
#pragma unroll
        for (int j = 0; j < 8; ++j) out[j] = (_Float16)0.f;
    }
    *(v8h*)(dst + (size_t)r * Kpad + k8) = out;
}

// ---------------------------------------------------------------------------
// C[M][N] = A[M][K] * B[N][K]^T, f16 in / f32 WMMA accumulate.
// Block tile 128x128, 8 waves (4x2), wave tile 32x64 = 2x4 WMMA tiles.
// Triple-buffered LDS fed by GLOBAL_LOAD_ASYNC_TO_LDS_B128: copies for
// chunk c+2 are issued in iteration c and only waited one iteration later
// (s_wait_asynccnt <= 4), giving a two-iteration overlap window.
// ---------------------------------------------------------------------------
template<int N, int K>
__global__ __launch_bounds__(256)
void snn_gemm_f16_wmma(const _Float16* __restrict__ Ag,
                       const _Float16* __restrict__ Bg,
                       float* __restrict__ C)
{
    __shared__ __align__(16) _Float16 As[3][BUFH];
    __shared__ __align__(16) _Float16 Bs[3][BUFH];

    const int tid  = threadIdx.x;
    const int lane = tid & 31;
    const int wave = tid >> 5;      // 0..7
    const int wm   = wave >> 1;     // 0..3 (M)
    const int wn   = wave & 1;      // 0..1 (N)
    const int half = lane >> 4;     // K-half held by this lane
    const int lr   = lane & 15;     // row (A) / col (B) within 16

    const int m_blk = blockIdx.y * BM;
    const int n_blk = blockIdx.x * BN;

    // Staging split (both A and B tiles are 128 rows x 32 halves):
    // 2 threads per row, 16 halves (= 2 async b128) each.
    const int srow  = tid >> 1;         // 0..127
    const int skofs = (tid & 1) * 16;   // 0 / 16

    const _Float16* Arow = Ag + (size_t)(m_blk + srow) * K + skofs;
    const _Float16* Brow = Bg + (size_t)(n_blk + srow) * K + skofs;

    // wave-relative LDS byte addresses of this thread's staging slot (buf 0)
    const unsigned slotA = (unsigned)(size_t)&As[0][srow * LDA + skofs];
    const unsigned slotB = (unsigned)(size_t)&Bs[0][srow * LDA + skofs];

    v8f acc[8] = {};   // [im][in_] flattened: acc[im*4 + in_]

    // prologue: async-stage chunk 0 -> buf0 and chunk 1 -> buf1
    async_g2l_b128(slotA,                 Arow);
    async_g2l_b128(slotA + 16,            Arow + 8);
    async_g2l_b128(slotB,                 Brow);
    async_g2l_b128(slotB + 16,            Brow + 8);
    async_g2l_b128(slotA + 10240,         Arow + BK);
    async_g2l_b128(slotA + 10240 + 16,    Arow + BK + 8);
    async_g2l_b128(slotB + 10240,         Brow + BK);
    async_g2l_b128(slotB + 10240 + 16,    Brow + BK + 8);
    wait_async_le4(acc);     // chunk 0 landed; chunk 1 may still fly
    __syncthreads();

    constexpr int NCH = K / BK;
    int cur = 0, nx2 = 2;    // compute buffer, issue buffer ((c+2)%3)

#pragma unroll 1
    for (int c = 0; c < NCH; ++c) {
        const bool issued = (c + 2 < NCH);
        if (issued) {        // async copies for chunk c+2 into buffer nx2
            const int k0n = (c + 2) * BK;
            const unsigned dA = slotA + (unsigned)nx2 * 10240;
            const unsigned dB = slotB + (unsigned)nx2 * 10240;
            async_g2l_b128(dA,      Arow + k0n);
            async_g2l_b128(dA + 16, Arow + k0n + 8);
            async_g2l_b128(dB,      Brow + k0n);
            async_g2l_b128(dB + 16, Brow + k0n + 8);
            if (c + 3 < NCH) {   // L2 prefetch one more chunk ahead
                __builtin_prefetch(Arow + (size_t)(c + 3) * BK, 0, 0);
                __builtin_prefetch(Brow + (size_t)(c + 3) * BK, 0, 0);
            }
        }

        // fragments from LDS (16-bit layout: lane<16 -> K0-7/16-23,
        // lane>=16 -> K8-15/24-31), then 2x4 WMMA
        const _Float16* curA = &As[0][0] + cur * BUFH;
        const _Float16* curB = &Bs[0][0] + cur * BUFH;
        Frag16 a[2], b[4];
#pragma unroll
        for (int im = 0; im < 2; ++im) {
            const int row = wm * 32 + im * 16 + lr;
            a[im].h[0] = *(const v8h*)(curA + row * LDA + half * 8);
            a[im].h[1] = *(const v8h*)(curA + row * LDA + 16 + half * 8);
        }
#pragma unroll
        for (int in_ = 0; in_ < 4; ++in_) {
            const int col = wn * 64 + in_ * 16 + lr;
            b[in_].h[0] = *(const v8h*)(curB + col * LDA + half * 8);
            b[in_].h[1] = *(const v8h*)(curB + col * LDA + 16 + half * 8);
        }
#pragma unroll
        for (int im = 0; im < 2; ++im)
#pragma unroll
            for (int in_ = 0; in_ < 4; ++in_)
                acc[im * 4 + in_] = __builtin_amdgcn_wmma_f32_16x16x32_f16(
                    false, a[im].v, false, b[in_].v, (short)0, acc[im * 4 + in_],
                    false, false);

        // chunk c+1 must have landed before anyone reads its buffer
        if (issued) wait_async_le4(acc);
        else        wait_async_le0(acc);
        __syncthreads();

        // rotate buffers: (cur, nx1, nx2) <- (nx1, nx2, cur)
        const int t = cur;
        cur = (cur == 2) ? 0 : cur + 1;
        nx2 = (nx2 == 2) ? 0 : nx2 + 1;
        (void)t;
    }

    // epilogue: D layout: lane = col, VGPR r -> row r + 8*half.
    // N is constexpr, so row strides fold into immediate store offsets.
#pragma unroll
    for (int im = 0; im < 2; ++im) {
#pragma unroll
        for (int in_ = 0; in_ < 4; ++in_) {
            const int gn  = n_blk + wn * 64 + in_ * 16 + lr;
            const int gm0 = m_blk + wm * 32 + im * 16 + half * 8;
            float* dst = C + (size_t)gm0 * N + gn;
#pragma unroll
            for (int r = 0; r < 8; ++r)
                dst[(size_t)r * N] = acc[im * 4 + in_][r];
        }
    }
}

// ---------------------------------------------------------------------------
// CUBA LIF: two (b,n) columns per thread, sequential over T.
// Spikes written as f16 (exact 0/1), packed b32 stores.
// ---------------------------------------------------------------------------
__global__ __launch_bounds__(256)
void snn_lif(const float* __restrict__ ch, _Float16* __restrict__ sh, int BN_, int T)
{
    const int idx2 = (blockIdx.x * blockDim.x + threadIdx.x) * 2;
    if (idx2 >= BN_) return;
    const float km = 1.0f / 6.0f;        // DT*TAU_MEM_INV
    const float kd = 1.0f - 1.0f / 6.0f; // 1 - DT*TAU_SYN_INV
    float v0 = 0.f, c0 = 0.f, v1 = 0.f, c1 = 0.f;
    for (int t = 0; t < T; ++t) {
        const float2 x = *(const float2*)(ch + (size_t)t * BN_ + idx2);
        v0 = v0 + km * (c0 - v0);  c0 = c0 * kd + x.x;
        v1 = v1 + km * (c1 - v1);  c1 = c1 * kd + x.y;
        const float z0 = (v0 - 1.0f > 0.f) ? 1.f : 0.f;
        const float z1 = (v1 - 1.0f > 0.f) ? 1.f : 0.f;
        v0 = (1.f - z0) * v0;
        v1 = (1.f - z1) * v1;
        v2h z; z[0] = (_Float16)z0; z[1] = (_Float16)z1;
        *(v2h*)(sh + (size_t)t * BN_ + idx2) = z;
    }
}

// CUBA LI readout, two columns per thread, in place on d_out
__global__ __launch_bounds__(256)
void snn_li(float* __restrict__ co, int BN_, int T)
{
    const int idx2 = (blockIdx.x * blockDim.x + threadIdx.x) * 2;
    if (idx2 >= BN_) return;
    const float km = 1.0f / 6.0f;
    const float kd = 1.0f - 1.0f / 6.0f;
    float v0 = 0.f, c0 = 0.f, v1 = 0.f, c1 = 0.f;
    for (int t = 0; t < T; ++t) {
        float2 x = *(const float2*)(co + (size_t)t * BN_ + idx2);
        v0 = v0 + km * (c0 - v0);  c0 = c0 * kd + x.x;
        v1 = v1 + km * (c1 - v1);  c1 = c1 * kd + x.y;
        float2 o; o.x = v0; o.y = v1;
        *(float2*)(co + (size_t)t * BN_ + idx2) = o;
    }
}

extern "C" void kernel_launch(void* const* d_in, const int* in_sizes, int n_in,
                              void* d_out, int out_size, void* d_ws, size_t ws_size,
                              hipStream_t stream)
{
    const float* spikes = (const float*)d_in[0];  // [256,128,784]
    const float* w_h    = (const float*)d_in[1];  // [512,784]
    const float* w_o    = (const float*)d_in[2];  // [128,512]
    float* out = (float*)d_out;                   // [256,128,128]

    const int T = 256, B = 128, NIN = 784;
    constexpr int NH = 512, NO = 128, KP1 = 800;  // 784 padded to 800
    const int M = T * B;                          // 32768

    // workspace layout
    char* ws = (char*)d_ws;
    size_t off = 0;
    float*    c_h   = (float*)(ws + off);    off += (size_t)M * NH * 4;   // 64 MB
    _Float16* s_h   = (_Float16*)(ws + off); off += (size_t)M * NH * 2;   // 32 MB
    _Float16* spk_h = (_Float16*)(ws + off); off += (size_t)M * KP1 * 2;  // 50 MB
    _Float16* w_h_h = (_Float16*)(ws + off); off += (size_t)NH * KP1 * 2; // 0.8 MB
    _Float16* w_o_h = (_Float16*)(ws + off);                              // 0.13 MB

    // f32 -> f16 (pad K to multiple of 32 with zeros)
    {
        int n1 = M * (KP1 >> 3);
        snn_cvt_f16_pad<<<(n1 + 255) / 256, 256, 0, stream>>>(spikes, spk_h, M, NIN, KP1);
        int n2 = NH * (KP1 >> 3);
        snn_cvt_f16_pad<<<(n2 + 255) / 256, 256, 0, stream>>>(w_h, w_h_h, NH, NIN, KP1);
        int n3 = NO * (NH >> 3);
        snn_cvt_f16_pad<<<(n3 + 255) / 256, 256, 0, stream>>>(w_o, w_o_h, NO, NH, NH);
    }

    // Layer 1 GEMM: c_h[tb][h] = sum_i spk[tb][i] * w_h[h][i]
    dim3 g1(NH / BN, M / BM);
    snn_gemm_f16_wmma<NH, KP1><<<g1, 256, 0, stream>>>(spk_h, w_h_h, c_h);

    // Hidden LIF scan -> spikes (f16, exact)
    snn_lif<<<(B * NH / 2) / 256, 256, 0, stream>>>(c_h, s_h, B * NH, T);

    // Layer 2 GEMM: c_o[tb][o] = sum_h s_h[tb][h] * w_o[o][h] -> d_out
    dim3 g2(NO / BN, M / BM);
    snn_gemm_f16_wmma<NO, NH><<<g2, 256, 0, stream>>>(s_h, w_o_h, out);

    // Readout LI scan, in place on d_out
    snn_li<<<(B * NO / 2) / 256, 256, 0, stream>>>(out, B * NO, T);
}
